// Net_91122026151953
// MI455X (gfx1250) — compile-verified, matrix operand order checked by reading, not run on hardware
//
#include <hip/hip_runtime.h>

// ---------------- problem constants ----------------
#define HID 180          // hidden size
#define TT  240          // timesteps
#define NBAT 2048        // batch
#define KP  192          // K padded to multiple of 32 for enc/dec GEMMs
#define KSE 6            // K-steps (192/32) enc/dec
#define NTE 45           // N-tiles (720/16) enc/dec
#define LDA 208          // LDS A-tile row stride (halfs), 16B-aligned, bank-skewed
#define GST 724          // LDS gate-buffer row stride (floats), bank-skewed
#define KS3 1350         // K-steps for output GEMM (43200/32)
#define NT3 3            // N-tiles for output GEMM (48/16, cols 40..47 are pad)

typedef __attribute__((ext_vector_type(16))) _Float16 v16h;
typedef __attribute__((ext_vector_type(8)))  _Float16 v8h;
typedef __attribute__((ext_vector_type(8)))  float    v8f;

// 16-bit A/B fragment K-mapping per CDNA5 ISA (16-bit A 16x32):
// lanes 0-15 : halfs 0-7 -> K 0-7,  halfs 8-15 -> K 16-23
// lanes 16-31: halfs 0-7 -> K 8-15, halfs 8-15 -> K 24-31
__device__ __forceinline__ int kmap(int lane, int h) {
  int base = (h < 8) ? h : (h + 8);
  return base + ((lane >= 16) ? 8 : 0);
}
// inverse: element (row/col m, kk in [0,32)) -> (lane, half)
__device__ __forceinline__ void kinv(int kk, int m, int& lane, int& h) {
  int hi = (kk >> 3) & 1;
  lane = m + (hi ? 16 : 0);
  h = kk - ((kk >= 16) ? 8 : 0) - (hi ? 8 : 0);
}

__device__ __forceinline__ float sigmoidf(float x) {
  return 1.0f / (1.0f + __expf(-x));
}

// load A fragment (per-lane 16 halfs) from an LDS row-major f16 tile
__device__ __forceinline__ v16h ld_frag_lds(const _Float16* p) {
  v8h lo = *(const v8h*)(p);        // halfs [0,8)
  v8h hi = *(const v8h*)(p + 16);   // halfs [16,24)
  v16h r;
#pragma unroll
  for (int i = 0; i < 8; ++i) { r[i] = lo[i]; r[8 + i] = hi[i]; }
  return r;
}

// ---------------- weight packing: fp32 [nrows,K] -> WMMA B fragments f16 ----------------
// layout: Bp[ ((nt*ksteps + ks)*32 + lane)*16 + h ] = W[nt*16 + lane%16][ks*32 + kmap(lane,h)]
__global__ __launch_bounds__(256) void pack_b_kernel(
    const float* __restrict__ W, _Float16* __restrict__ Bp,
    int nrows, int K, int ksteps, int ntiles)
{
  size_t idx = (size_t)blockIdx.x * 256 + threadIdx.x;
  size_t total = (size_t)ntiles * ksteps * 512;
  if (idx >= total) return;
  int h    = (int)(idx & 15);
  int lane = (int)((idx >> 4) & 31);
  size_t rest = idx >> 9;
  int ks = (int)(rest % (size_t)ksteps);
  int nt = (int)(rest / (size_t)ksteps);
  int n = nt * 16 + (lane & 15);
  int k = ks * 32 + kmap(lane, h);
  float v = 0.0f;
  if (n < nrows && k < K) v = W[(size_t)n * K + k];
  Bp[idx] = (_Float16)v;
}

// ---------------- encoder: gates GEMM + LSTM-cell(zero state) + softmax ----------------
// one WG per (b, t-tile of 16): M-tile = 16 rows, N = 720 (forget-gate tiles skipped)
__global__ __launch_bounds__(256) void enc_kernel(
    const float* __restrict__ x,
    const _Float16* __restrict__ Bpack,
    const float* __restrict__ bih, const float* __restrict__ bhh,
    _Float16* __restrict__ Aenc)
{
  __shared__ _Float16 lds_a[16 * LDA];   // 6.5 KB  f16 A tile (K padded)
  __shared__ float    lds_g[16 * GST];   // 46.3 KB f32 gate tile

  int wg = blockIdx.x;                   // 0..30719
  int b  = wg / 15;
  int t0 = (wg % 15) * 16;
  int tid  = threadIdx.x;
  int lane = tid & 31;
  int wave = tid >> 5;

  // stage xe[b, t0..t0+15, 0..179] -> LDS f16 (x layout: [b][c][f][t], k=c*60+f)
  {
    int r = tid & 15;                    // t within tile -> contiguous in memory
    const float* xb = x + (size_t)b * (3 * 60 * TT) + t0 + r;
    for (int k = tid >> 4; k < HID; k += 16)
      lds_a[r * LDA + k] = (_Float16)xb[(size_t)k * TT];
    for (int k = HID + (tid >> 4); k < KP; k += 16)
      lds_a[r * LDA + k] = (_Float16)0.0f;
  }
  __syncthreads();

  const v16h* Bp = (const v16h*)Bpack;
  for (int nt = wave; nt < NTE; nt += 8) {
    if (nt >= 12 && nt <= 21) continue;  // forget gate * c_prev(=0): dead columns
    v8f acc = {};
#pragma unroll
    for (int ks = 0; ks < KSE; ++ks) {
      int off = (lane < 16) ? 0 : 8;
      v16h a  = ld_frag_lds(&lds_a[(lane & 15) * LDA + ks * 32 + off]);
      v16h bf = Bp[(size_t)(nt * KSE + ks) * 32 + lane];
      acc = __builtin_amdgcn_wmma_f32_16x16x32_f16(false, a, false, bf,
                                                   (short)0, acc, false, false);
    }
    int n = lane & 15;
    int g = nt * 16 + n;
    float bias = bih[g] + bhh[g];
    int mbase = (lane >> 4) * 8;         // C layout: vgpr r -> row r (+8 for hi lanes)
#pragma unroll
    for (int r = 0; r < 8; ++r)
      lds_g[(mbase + r) * GST + g] = acc[r] + bias;
  }
  __syncthreads();

  // activations + softmax over H; each half-wave owns one row
  int r  = 2 * wave + (lane >> 4);
  int j0 = lane & 15;
  float hv[12];
  float mx = -1e30f;
#pragma unroll
  for (int i = 0; i < 12; ++i) {
    int j = j0 + 16 * i;
    float v = -1e30f;
    if (j < HID) {
      float gi = lds_g[r * GST + j];
      float gg = lds_g[r * GST + 360 + j];
      float go = lds_g[r * GST + 540 + j];
      float c  = sigmoidf(gi) * tanhf(gg);
      v = sigmoidf(go) * tanhf(c);
    }
    hv[i] = v;
    mx = fmaxf(mx, v);
  }
#pragma unroll
  for (int m = 1; m < 16; m <<= 1) mx = fmaxf(mx, __shfl_xor(mx, m, 32));
  float s = 0.0f;
#pragma unroll
  for (int i = 0; i < 12; ++i) {
    int j = j0 + 16 * i;
    if (j < HID) { float e = __expf(hv[i] - mx); hv[i] = e; s += e; }
  }
#pragma unroll
  for (int m = 1; m < 16; m <<= 1) s += __shfl_xor(s, m, 32);
  float inv = 1.0f / s;

  // write enc tile directly in WMMA A-fragment layout for the decoder GEMM
  _Float16* At = Aenc + (size_t)wg * (KSE * 512);
#pragma unroll
  for (int i = 0; i < 12; ++i) {
    int j = j0 + 16 * i;
    if (j < HID) {
      int ks = j >> 5, kk = j & 31, l2, h2;
      kinv(kk, r, l2, h2);
      At[(size_t)ks * 512 + l2 * 16 + h2] = (_Float16)(hv[i] * inv);
    }
  }
  if (j0 < 12) {                         // zero-pad K 180..191 (ks=5, kk=20..31)
    int kk = 20 + j0, l2, h2;
    kinv(kk, r, l2, h2);
    At[(size_t)5 * 512 + l2 * 16 + h2] = (_Float16)0.0f;
  }
}

// ---------------- decoder: gates GEMM + LSTM-cell, repack h_dec for output GEMM ----------------
__global__ __launch_bounds__(256) void dec_kernel(
    const _Float16* __restrict__ Aenc,
    const _Float16* __restrict__ Bpack,
    const float* __restrict__ bih, const float* __restrict__ bhh,
    _Float16* __restrict__ Aout)
{
  __shared__ float lds_g[16 * GST];
  int wg = blockIdx.x;
  int b  = wg / 15;
  int t0 = (wg % 15) * 16;
  int tid  = threadIdx.x;
  int lane = tid & 31;
  int wave = tid >> 5;

  const v16h* Ap = (const v16h*)(Aenc + (size_t)wg * (KSE * 512));
  const v16h* Bp = (const v16h*)Bpack;
  for (int nt = wave; nt < NTE; nt += 8) {
    if (nt >= 12 && nt <= 21) continue;
    v8f acc = {};
#pragma unroll
    for (int ks = 0; ks < KSE; ++ks) {
      v16h a  = Ap[(size_t)ks * 32 + lane];
      v16h bf = Bp[(size_t)(nt * KSE + ks) * 32 + lane];
      acc = __builtin_amdgcn_wmma_f32_16x16x32_f16(false, a, false, bf,
                                                   (short)0, acc, false, false);
    }
    int n = lane & 15;
    int g = nt * 16 + n;
    float bias = bih[g] + bhh[g];
    int mbase = (lane >> 4) * 8;
#pragma unroll
    for (int r = 0; r < 8; ++r)
      lds_g[(mbase + r) * GST + g] = acc[r] + bias;
  }
  __syncthreads();

  int r  = 2 * wave + (lane >> 4);
  int j0 = lane & 15;
  int m  = b & 15;                       // row within output-GEMM M-tile
  size_t tb = (size_t)(b >> 4) * KS3 * 512;
  int t = t0 + r;
#pragma unroll
  for (int i = 0; i < 12; ++i) {
    int j = j0 + 16 * i;
    if (j < HID) {
      float gi = lds_g[r * GST + j];
      float gg = lds_g[r * GST + 360 + j];
      float go = lds_g[r * GST + 540 + j];
      float c  = sigmoidf(gi) * tanhf(gg);
      float hval = sigmoidf(go) * tanhf(c);
      int k = t * HID + j;               // flat K index 0..43199 (multiple-of-32 total)
      int ks = k >> 5, kk = k & 31, l2, h2;
      kinv(kk, m, l2, h2);
      Aout[tb + (size_t)ks * 512 + l2 * 16 + h2] = (_Float16)hval;
    }
  }
}

// ---------------- output head: [2048,43200]x[43200,48] + group softmax ----------------
__global__ __launch_bounds__(256) void out_kernel(
    const _Float16* __restrict__ Aout,
    const _Float16* __restrict__ Bpack,
    const float* __restrict__ bout,
    float* __restrict__ out)
{
  __shared__ float lds_c[8 * 16 * 48];   // per-wave partial C tiles, 24.6 KB
  int mtile = blockIdx.x;                // 0..127
  int tid  = threadIdx.x;
  int lane = tid & 31;
  int wave = tid >> 5;

  const v16h* Ap = (const v16h*)Aout + (size_t)mtile * KS3 * 32;
  const v16h* Bp = (const v16h*)Bpack;
  v8f a0 = {}, a1 = {}, a2 = {};
  for (int ks = wave; ks < KS3; ks += 8) {       // K split across 8 waves
    v16h a  = Ap[(size_t)ks * 32 + lane];
    v16h b0 = Bp[(size_t)(0 * KS3 + ks) * 32 + lane];
    v16h b1 = Bp[(size_t)(1 * KS3 + ks) * 32 + lane];
    v16h b2 = Bp[(size_t)(2 * KS3 + ks) * 32 + lane];
    a0 = __builtin_amdgcn_wmma_f32_16x16x32_f16(false, a, false, b0, (short)0, a0, false, false);
    a1 = __builtin_amdgcn_wmma_f32_16x16x32_f16(false, a, false, b1, (short)0, a1, false, false);
    a2 = __builtin_amdgcn_wmma_f32_16x16x32_f16(false, a, false, b2, (short)0, a2, false, false);
  }
  int n = lane & 15, mbase = (lane >> 4) * 8;
#pragma unroll
  for (int r = 0; r < 8; ++r) {
    float* row = &lds_c[(wave * 16 + mbase + r) * 48];
    row[n] = a0[r]; row[16 + n] = a1[r]; row[32 + n] = a2[r];
  }
  __syncthreads();

  if (tid < 64) {                                // 16 rows x 4 softmax groups of 10
    int row = tid >> 2, grp = tid & 3;
    float v[10]; float mx = -1e30f;
#pragma unroll
    for (int q = 0; q < 10; ++q) {
      int col = grp * 10 + q;
      float s = bout[col];
      for (int w = 0; w < 8; ++w) s += lds_c[(w * 16 + row) * 48 + col];
      v[q] = s; mx = fmaxf(mx, s);
    }
    float sum = 0.0f;
#pragma unroll
    for (int q = 0; q < 10; ++q) { float e = __expf(v[q] - mx); v[q] = e; sum += e; }
    float inv = 1.0f / sum;
    float* op = out + (size_t)(mtile * 16 + row) * 40 + grp * 10;
#pragma unroll
    for (int q = 0; q < 10; ++q) op[q] = v[q] * inv;
  }
}

// ---------------- launch ----------------
extern "C" void kernel_launch(void* const* d_in, const int* in_sizes, int n_in,
                              void* d_out, int out_size, void* d_ws, size_t ws_size,
                              hipStream_t stream)
{
  (void)in_sizes; (void)n_in; (void)out_size; (void)ws_size;
  const float* x    = (const float*)d_in[0];
  const float* Wenc = (const float*)d_in[1];
  const float* bihE = (const float*)d_in[2];
  const float* bhhE = (const float*)d_in[3];
  const float* Wdec = (const float*)d_in[4];
  const float* bihD = (const float*)d_in[5];
  const float* bhhD = (const float*)d_in[6];
  const float* Wout = (const float*)d_in[7];
  const float* bout = (const float*)d_in[8];
  float* out = (float*)d_out;

  // workspace layout (all offsets 256B aligned):
  //   Benc 276480 | Bdec 276480 | Bout 4147200 | Aenc 188743680 | Aout 176947200
  char* ws = (char*)d_ws;
  _Float16* Benc = (_Float16*)(ws + 0);
  _Float16* Bdec = (_Float16*)(ws + 276480);
  _Float16* Bout = (_Float16*)(ws + 552960);
  _Float16* Aenc = (_Float16*)(ws + 4700160);
  _Float16* Aout = (_Float16*)(ws + 4700160 + (size_t)188743680);

  pack_b_kernel<<<(NTE * KSE * 512) / 256, 256, 0, stream>>>(Wenc, Benc, 720, HID, KSE, NTE);
  pack_b_kernel<<<(NTE * KSE * 512) / 256, 256, 0, stream>>>(Wdec, Bdec, 720, HID, KSE, NTE);
  pack_b_kernel<<<(NT3 * KS3 * 512) / 256, 256, 0, stream>>>(Wout, Bout, 40, TT * HID, KS3, NT3);

  enc_kernel<<<NBAT * 15, 256, 0, stream>>>(x, Benc, bihE, bhhE, Aenc);
  dec_kernel<<<NBAT * 15, 256, 0, stream>>>(Aenc, Bdec, bihD, bhhD, Aout);
  out_kernel<<<NBAT / 16, 256, 0, stream>>>(Aout, Bout, bout, out);
}
